// FeatureDecouplingLoss_72533407695362
// MI455X (gfx1250) — matrix operand order repaired and checked by reading, not compiled, not run on hardware
//
#include <hip/hip_runtime.h>
#include <hip/hip_bf16.h>

// ---------------------------------------------------------------------------
// FeatureDecouplingLoss on MI455X (gfx1250, wave32, WMMA)
//
// Only same-class pairs contribute: compute per-class Gram matrices of the
// L2-normalized rows with v_wmma_f32_16x16x32_f16 (f16 in, f32 accumulate).
// Rows are permuted into class-sorted, 16-padded "position" space; padding
// slots are zero so WMMA tiles need no bounds checks (EXEC stays all-ones).
// A-fragments read the row-major packed matrix pn16[pos][d]; B-fragments
// read the same matrix pn16[pos][d] (lane L holds B column N=L%16 as 16
// halves in two runs of 8: k = 8h.. and 16+8h..). All class data sits in the 192MB L2.
// ---------------------------------------------------------------------------

typedef __attribute__((ext_vector_type(16))) _Float16 v16h;
typedef __attribute__((ext_vector_type(8)))  _Float16 v8h;
typedef __attribute__((ext_vector_type(8)))  float    v8f;

#define B_ROWS   8192
#define DIMS     1024
#define NCLS     64
#define PMAX     9152      // 8192 + 64*15 rounded to 16: max padded positions (286*32)
#define DESC_CAP 262144    // worst case: one class owns all rows -> (8192/16)^2

// ---- workspace layout (bytes) ----
#define O_LOSS    0              // float
#define O_NUNIQ   4              // int
#define O_NTILES  8              // int
#define O_COUNTS  64             // int[64]
#define O_CURSOR  320            // int[64]
#define O_POFF    576            // int[64]
#define O_INV     1024           // float[8192]
#define O_POSMAP  33792          // int[PMAX]
#define O_DESC    70656          // int4[DESC_CAP]  (4 MB)
#define O_PN      4264960        // half[PMAX*DIMS] (18.7 MB)
#define O_PNT     23008256       // half[DIMS*PMAX] (18.7 MB)
// total ~41.8 MB

__global__ void fdl_init(float* loss, int* nuniq, int* ntiles,
                         int* counts, int* cursor, int* posmap) {
  int i = blockIdx.x * blockDim.x + threadIdx.x;
  if (i == 0) { *loss = 0.0f; *nuniq = 0; *ntiles = 0; }
  if (i < NCLS) { counts[i] = 0; cursor[i] = 0; }
  if (i < PMAX) posmap[i] = -1;
}

__global__ void fdl_count(const int* __restrict__ labels, int* __restrict__ counts) {
  int i = blockIdx.x * blockDim.x + threadIdx.x;
  if (i < B_ROWS) atomicAdd(&counts[labels[i] & (NCLS - 1)], 1);
}

// Single block of 64 threads: exclusive scans over padded sizes and tile
// counts, unique-class count, and tile-descriptor emission.
__global__ void fdl_scan_emit(const int* __restrict__ counts, int* __restrict__ poff,
                              int* __restrict__ nuniq, int* __restrict__ ntiles,
                              int4* __restrict__ desc) {
  __shared__ int sN[NCLS], sPoff[NCLS], sToff[NCLS];
  int c = threadIdx.x;
  int n = counts[c];
  sN[c] = n;
  __syncthreads();
  if (c == 0) {
    int accP = 0, accT = 0, u = 0;
    for (int k = 0; k < NCLS; ++k) {
      int T = (sN[k] + 15) >> 4;
      sPoff[k] = accP; accP += T << 4;
      sToff[k] = accT; accT += T * T;
      if (sN[k] > 0) ++u;
    }
    *ntiles = accT;
    *nuniq = u;
  }
  __syncthreads();
  poff[c] = sPoff[c];
  int T = (n + 15) >> 4;
  int base = sToff[c], p0 = sPoff[c];
  for (int ti = 0; ti < T; ++ti)
    for (int tj = 0; tj < T; ++tj)
      desc[base + ti * T + tj] = make_int4(p0 + (ti << 4), p0 + (tj << 4), n, 0);
}

__global__ void fdl_place(const int* __restrict__ labels, const int* __restrict__ poff,
                          int* __restrict__ cursor, int* __restrict__ posmap) {
  int i = blockIdx.x * blockDim.x + threadIdx.x;
  if (i < B_ROWS) {
    int c = labels[i] & (NCLS - 1);
    int r = atomicAdd(&cursor[c], 1);
    posmap[poff[c] + r] = i;
  }
}

// One block per row: 1/max(||row||, eps)
__global__ void fdl_norm(const float* __restrict__ feat, float* __restrict__ inv_norm) {
  __shared__ float red[256];
  int row = blockIdx.x;
  float ss = 0.0f;
  for (int d = threadIdx.x; d < DIMS; d += 256) {
    float v = feat[(size_t)row * DIMS + d];
    ss += v * v;
  }
  red[threadIdx.x] = ss;
  __syncthreads();
  for (int off = 128; off > 0; off >>= 1) {
    if (threadIdx.x < off) red[threadIdx.x] += red[threadIdx.x + off];
    __syncthreads();
  }
  if (threadIdx.x == 0) inv_norm[row] = 1.0f / fmaxf(sqrtf(red[0]), 1e-12f);
}

// 32x32 LDS-tiled normalize + f16 convert, emitting both pn16 [pos][d]
// (A- and B-fragment feed) and pnT16 [d][pos] (not read by fdl_gram). Unmapped (padding)
// positions are written as zeros.
__global__ void fdl_fill(const float* __restrict__ feat, const float* __restrict__ inv_norm,
                         const int* __restrict__ posmap,
                         _Float16* __restrict__ pn, _Float16* __restrict__ pnt) {
  __shared__ _Float16 tile[32][33];
  const int tx = threadIdx.x;          // 0..31
  const int ty = threadIdx.y;          // 0..7
  const int pos0 = blockIdx.x << 5;
  const int d0   = blockIdx.y << 5;
  #pragma unroll
  for (int i = 0; i < 4; ++i) {
    int pr = ty + (i << 3);
    int row = posmap[pos0 + pr];
    float v = 0.0f;
    if (row >= 0) v = feat[(size_t)row * DIMS + d0 + tx] * inv_norm[row];
    tile[pr][tx] = (_Float16)v;
  }
  __syncthreads();
  #pragma unroll
  for (int i = 0; i < 4; ++i) {
    int pr = ty + (i << 3);
    pn[(size_t)(pos0 + pr) * DIMS + d0 + tx] = tile[pr][tx];
  }
  #pragma unroll
  for (int i = 0; i < 4; ++i) {
    int dr = ty + (i << 3);
    pnt[(size_t)(d0 + dr) * PMAX + pos0 + tx] = tile[tx][dr];
  }
}

// Persistent waves over 16x16 class tiles; K=1024 in 32 steps of
// v_wmma_f32_16x16x32_f16. Diagonal (i==j) excluded by position compare.
__global__ void fdl_gram(const _Float16* __restrict__ pn, const _Float16* __restrict__ pnt,
                         const int4* __restrict__ desc, const int* __restrict__ ntiles,
                         float* __restrict__ loss_acc) {
  const int lane  = threadIdx.x & 31;
  const int wid   = (blockIdx.x * blockDim.x + threadIdx.x) >> 5;
  const int nwave = (gridDim.x * blockDim.x) >> 5;
  const int total = *ntiles;
  const int nidx  = lane & 15;            // A row / C column within tile
  const int mbase = (lane >> 4) << 3;     // C row base (0 or 8)
  const int akoff = (lane >> 4) << 3;     // A K-offset in halves (0 or 8)

  float local = 0.0f;
  for (int t = wid; t < total; t += nwave) {
    int4 dd = desc[t];
    const int p0 = dd.x, q0 = dd.y, nc = dd.z;
    const _Float16* arow = pn  + (size_t)(p0 + nidx) * DIMS + akoff;
    const _Float16* bcol = pn  + (size_t)(q0 + nidx) * DIMS + akoff;   // B column N = nidx
    v8f acc = {};
    #pragma unroll 4
    for (int kb = 0; kb < DIMS; kb += 32) {
      v8h alo = *(const v8h*)(arow + kb);        // K = akoff+kb .. +7
      v8h ahi = *(const v8h*)(arow + kb + 16);   // K = akoff+kb+16 .. +23
      const _Float16* bp = bcol + kb;
      v8h blo = *(const v8h*)(bp);               // K = akoff+kb .. +7
      v8h bhi = *(const v8h*)(bp + 16);          // K = akoff+kb+16 .. +23
      v16h a = __builtin_shufflevector(alo, ahi, 0,1,2,3,4,5,6,7,8,9,10,11,12,13,14,15);
      v16h b = __builtin_shufflevector(blo, bhi, 0,1,2,3,4,5,6,7,8,9,10,11,12,13,14,15);
      acc = __builtin_amdgcn_wmma_f32_16x16x32_f16(false, a, false, b,
                                                   (short)0, acc, false, false);
    }
    float s = 0.0f;
    #pragma unroll
    for (int e = 0; e < 8; ++e) {
      int pi = p0 + mbase + e;     // C element (M = mbase+e, N = nidx)
      int pj = q0 + nidx;
      float v = acc[e];
      s += (pi != pj) ? v * v : 0.0f;
    }
    local += s / ((float)nc * (float)DIMS);
  }

  __shared__ float red[128];
  red[threadIdx.x] = local;
  __syncthreads();
  for (int off = 64; off > 0; off >>= 1) {
    if (threadIdx.x < off) red[threadIdx.x] += red[threadIdx.x + off];
    __syncthreads();
  }
  if (threadIdx.x == 0) atomicAdd(loss_acc, red[0]);
}

__global__ void fdl_finish(const float* __restrict__ loss, const int* __restrict__ nuniq,
                           float* __restrict__ out) {
  if (blockIdx.x == 0 && threadIdx.x == 0) {
    int u = *nuniq;
    out[0] = *loss / (float)(u > 0 ? u : 1);
  }
}

extern "C" void kernel_launch(void* const* d_in, const int* in_sizes, int n_in,
                              void* d_out, int out_size, void* d_ws, size_t ws_size,
                              hipStream_t stream) {
  const float* feat  = (const float*)d_in[0];
  const int*   label = (const int*)d_in[1];
  float* out = (float*)d_out;

  char* w = (char*)d_ws;
  float* loss   = (float*)(w + O_LOSS);
  int*   nuniq  = (int*)  (w + O_NUNIQ);
  int*   ntiles = (int*)  (w + O_NTILES);
  int*   counts = (int*)  (w + O_COUNTS);
  int*   cursor = (int*)  (w + O_CURSOR);
  int*   poff   = (int*)  (w + O_POFF);
  float* invn   = (float*)(w + O_INV);
  int*   posmap = (int*)  (w + O_POSMAP);
  int4*  desc   = (int4*) (w + O_DESC);
  _Float16* pn  = (_Float16*)(w + O_PN);
  _Float16* pnt = (_Float16*)(w + O_PNT);

  (void)in_sizes; (void)n_in; (void)out_size; (void)ws_size;

  fdl_init<<<(PMAX + 255) / 256, 256, 0, stream>>>(loss, nuniq, ntiles, counts, cursor, posmap);
  fdl_count<<<B_ROWS / 256, 256, 0, stream>>>(label, counts);
  fdl_scan_emit<<<1, NCLS, 0, stream>>>(counts, poff, nuniq, ntiles, desc);
  fdl_place<<<B_ROWS / 256, 256, 0, stream>>>(label, poff, cursor, posmap);
  fdl_norm<<<B_ROWS, 256, 0, stream>>>(feat, invn);
  {
    dim3 grid(PMAX / 32, DIMS / 32);
    dim3 blk(32, 8);
    fdl_fill<<<grid, blk, 0, stream>>>(feat, invn, posmap, pn, pnt);
  }
  fdl_gram<<<256, 128, 0, stream>>>(pn, pnt, desc, ntiles, loss);
  fdl_finish<<<1, 1, 0, stream>>>(loss, nuniq, out);
}
